// GCN_47047071760480
// MI455X (gfx1250) — compile-verified, matrix operand order checked by reading, not run on hardware
//
#include <hip/hip_runtime.h>

typedef __attribute__((ext_vector_type(16))) _Float16 v16h;
typedef __attribute__((ext_vector_type(8)))  float    v8f;

#define NNODES 50000
#define NEDGES 625000
#define DFEAT  128

// ---------------------------------------------------------------------------
// small utility kernels
// ---------------------------------------------------------------------------
__global__ void fill_zero_f32(float* __restrict__ p, int n) {
  int i = blockIdx.x * blockDim.x + threadIdx.x;
  if (i < n) p[i] = 0.0f;
}

__global__ void degree_accum(const int* __restrict__ dst, float* __restrict__ deg, int nE) {
  int e = blockIdx.x * blockDim.x + threadIdx.x;
  if (e < nE) atomicAdd(deg + dst[e], 1.0f);
}

__global__ void deg_to_rsqrt(float* __restrict__ dinv, int n) {
  int i = blockIdx.x * blockDim.x + threadIdx.x;
  if (i < n) dinv[i] = __frsqrt_rn(dinv[i] + 1.0f);
}

// ---------------------------------------------------------------------------
// WMMA GEMM: Y[M x 128] = A[M x 128] (f32, cast to f16) @ W[128 x 128]
// One wave per 16-row strip; 8 accumulator tiles cover all 128 columns.
// W is staged into LDS in f16, pre-swizzled into the WMMA B-operand layout
// so each B tile fragment is one contiguous 32B LDS read per lane.
// ---------------------------------------------------------------------------
__global__ void __launch_bounds__(128)
gemm_wmma_f16(const float* __restrict__ A, const float* __restrict__ W,
              float* __restrict__ Y, int M) {
  __shared__ __align__(32) _Float16 Wl[128 * 128];  // 32 KB of 320 KB WGP LDS

  // Swizzled fill: element index i = (((ks*8 + nt)*32 + lane)*16 + j)
  // maps to W[k][n] with the 16-bit B-fragment layout (kg = lane>>4).
  for (int i = threadIdx.x; i < 128 * 128; i += blockDim.x) {
    int j    = i & 15;
    int lane = (i >> 4) & 31;
    int t    = i >> 9;          // 0..31 == ks*8 + nt
    int ks   = t >> 3;
    int nt   = t & 7;
    int kg   = lane >> 4;
    int n    = (nt << 4) + (lane & 15);
    int k    = (ks << 5) + ((j < 8) ? (j + (kg << 3)) : (16 + (j - 8) + (kg << 3)));
    Wl[i] = (_Float16)W[k * 128 + n];
  }
  __syncthreads();

  const int wave  = threadIdx.x >> 5;
  const int lane  = threadIdx.x & 31;
  const int mtile = blockIdx.x * 4 + wave;
  if (mtile * 16 >= M) return;               // wave-uniform: EXEC stays all-1s

  const int m  = lane & 15;
  const int kg = lane >> 4;
  const int row = mtile * 16 + m;

  v8f acc[8];
  v8f zero = {};
#pragma unroll
  for (int i = 0; i < 8; ++i) acc[i] = zero;

#pragma unroll
  for (int ks = 0; ks < 4; ++ks) {
    const int kbase = ks << 5;               // K step of 32
    // A fragment (16x32 f16): lane holds two 8-wide contiguous K runs
    const float4* ap = reinterpret_cast<const float4*>(A + (size_t)row * 128 + kbase + (kg << 3));
    float tmp[16];
    *(float4*)(tmp)      = ap[0];
    *(float4*)(tmp + 4)  = ap[1];
    *(float4*)(tmp + 8)  = ap[4];            // +16 floats
    *(float4*)(tmp + 12) = ap[5];
    v16h a;
#pragma unroll
    for (int j = 0; j < 16; ++j) a[j] = (_Float16)tmp[j];

#pragma unroll
    for (int nt = 0; nt < 8; ++nt) {
      const v16h b = *reinterpret_cast<const v16h*>(Wl + ((((ks << 3) + nt) << 5) + lane) * 16);
      acc[nt] = __builtin_amdgcn_wmma_f32_16x16x32_f16(
          /*neg_a=*/false, a, /*neg_b=*/false, b,
          /*c_mod=*/(short)0, acc[nt], /*reuse_a=*/false, /*reuse_b=*/false);
    }
  }

  // D layout: lane L -> n = L&15, row m = r + 8*(L>>4) for VGPR r
  const int n0    = lane & 15;
  const int mbase = mtile * 16 + (kg << 3);
#pragma unroll
  for (int nt = 0; nt < 8; ++nt) {
#pragma unroll
    for (int r = 0; r < 8; ++r) {
      Y[(size_t)(mbase + r) * 128 + (nt << 4) + n0] = acc[nt][r];
    }
  }
}

// ---------------------------------------------------------------------------
// Edge scatter: agg[dst] += (dinv[src]*dinv[dst]) * xw[src]
// 128 threads per edge (feature-coalesced); f32 global atomics into L2.
// ---------------------------------------------------------------------------
__global__ void edge_scatter(const int* __restrict__ src, const int* __restrict__ dst,
                             const float* __restrict__ dinv,
                             const float* __restrict__ xw,
                             float* __restrict__ agg, int nE) {
  unsigned int gid = blockIdx.x * blockDim.x + threadIdx.x;
  unsigned int e = gid >> 7;
  unsigned int f = gid & 127u;
  if (e >= (unsigned int)nE) return;
  int s = src[e], d = dst[e];
  float norm = dinv[s] * dinv[d];
  atomicAdd(agg + (size_t)d * DFEAT + f, norm * xw[(size_t)s * DFEAT + f]);
}

// ---------------------------------------------------------------------------
// epilogues: self-loop + bias + activation
// ---------------------------------------------------------------------------
__global__ void self_bias_relu(float* __restrict__ agg, const float* __restrict__ xw,
                               const float* __restrict__ dinv, const float* __restrict__ b,
                               int total) {
  int gid = blockIdx.x * blockDim.x + threadIdx.x;
  if (gid >= total) return;
  int i = gid >> 7;
  int f = gid & 127;
  float di = dinv[i];
  float v = agg[gid] + di * di * xw[gid] + b[f];
  agg[gid] = fmaxf(v, 0.0f);
}

__global__ void self_bias_sigmoid(float* __restrict__ out, const float* __restrict__ xw,
                                  const float* __restrict__ dinv, const float* __restrict__ b,
                                  int total) {
  int gid = blockIdx.x * blockDim.x + threadIdx.x;
  if (gid >= total) return;
  int i = gid >> 7;
  int f = gid & 127;
  float di = dinv[i];
  float v = out[gid] + di * di * xw[gid] + b[f];
  out[gid] = 1.0f / (1.0f + __expf(-v));
}

// ---------------------------------------------------------------------------
// launch
// ---------------------------------------------------------------------------
extern "C" void kernel_launch(void* const* d_in, const int* in_sizes, int n_in,
                              void* d_out, int out_size, void* d_ws, size_t ws_size,
                              hipStream_t stream) {
  (void)in_sizes; (void)n_in; (void)out_size; (void)ws_size;

  const float* x  = (const float*)d_in[0];
  const int*   ei = (const int*)d_in[1];
  const float* W1 = (const float*)d_in[2];
  const float* b1 = (const float*)d_in[3];
  const float* W2 = (const float*)d_in[4];
  const float* b2 = (const float*)d_in[5];
  float* out = (float*)d_out;

  const int* src = ei;            // edge_index[0]
  const int* dst = ei + NEDGES;   // edge_index[1]

  char*  ws   = (char*)d_ws;
  float* dinv = (float*)ws;                                        // 50000 f32
  float* bufA = (float*)(ws + 200192);                             // xw  : 6.4M f32
  float* bufB = (float*)(ws + 200192 + (size_t)NNODES * DFEAT * 4); // agg/h

  const int NF = NNODES * DFEAT;                 // 6,400,000
  const int nfBlocks   = (NF + 255) / 256;
  const int nodeBlocks = (NNODES + 255) / 256;
  const int edgeBlocks = (NEDGES + 255) / 256;
  const int scatBlocks = (int)(((size_t)NEDGES * DFEAT + 255) / 256);  // 312500
  const int gemmBlocks = (NNODES / 16 + 3) / 4;  // 3125 m-tiles, 4 waves/block

  // degree -> dinv (also zero d_out which doubles as layer-2 accumulator)
  fill_zero_f32<<<nodeBlocks, 256, 0, stream>>>(dinv, NNODES);
  fill_zero_f32<<<nfBlocks, 256, 0, stream>>>(out, NF);
  degree_accum<<<edgeBlocks, 256, 0, stream>>>(dst, dinv, NEDGES);
  deg_to_rsqrt<<<nodeBlocks, 256, 0, stream>>>(dinv, NNODES);

  // ---- layer 1 ----
  gemm_wmma_f16<<<gemmBlocks, 128, 0, stream>>>(x, W1, bufA, NNODES);
  fill_zero_f32<<<nfBlocks, 256, 0, stream>>>(bufB, NF);
  edge_scatter<<<scatBlocks, 256, 0, stream>>>(src, dst, dinv, bufA, bufB, NEDGES);
  self_bias_relu<<<nfBlocks, 256, 0, stream>>>(bufB, bufA, dinv, b1, NF);  // h in-place

  // ---- layer 2 ----
  gemm_wmma_f16<<<gemmBlocks, 128, 0, stream>>>(bufB, W2, bufA, NNODES);   // xw2 over bufA
  edge_scatter<<<scatBlocks, 256, 0, stream>>>(src, dst, dinv, bufA, out, NEDGES);
  self_bias_sigmoid<<<nfBlocks, 256, 0, stream>>>(out, bufA, dinv, b2, NF);
}
